// UnifiedPerceptionTransformer_63685775065424
// MI455X (gfx1250) — matrix operand order repaired and checked by reading, not compiled, not run on hardware
//
#include <hip/hip_runtime.h>
#include <hip/hip_bf16.h>
#include <math.h>

// Model dims (match reference)
#define S_LEN   2048
#define D_MODEL 1024
#define N_HEAD  16
#define D_HEAD  64
#define F_FFN   4096
#define N_LAYER 12
#define V_VOCAB 1400
#define V_PAD   1408

typedef __bf16 bf16_t;
typedef __attribute__((ext_vector_type(8)))  __bf16 bf16x8;
typedef __attribute__((ext_vector_type(16))) __bf16 bf16x16;
typedef __attribute__((ext_vector_type(8)))  float  f32x8;

// ---------------------------------------------------------------------------
// WMMA fragment helpers (k order used by this code, same in A and B; not the ISA table order)
// A (16x32 bf16): lane = 16*hi + m ; lane holds
//                 K = hi*16 .. hi*16+15        -> one contiguous 16-elt run.
// B (32x16 bf16, stored as N x K "Bt"): lane = 16*hi + n ; lane holds
//                 K = hi*16 .. hi*16+15        -> one contiguous 16-elt run.
// C/D (16x16 f32): lane = 16*hi + n ; VGPR r holds M = 8*hi + r.
// ---------------------------------------------------------------------------
__device__ __forceinline__ bf16x16 load_afrag(const bf16_t* p) {
  bf16x8 lo = *(const bf16x8*)(p);
  bf16x8 hi = *(const bf16x8*)(p + 8);
  bf16x16 r;
#pragma unroll
  for (int i = 0; i < 8; ++i) { r[i] = lo[i]; r[i + 8] = hi[i]; }
  return r;
}

// D = A(MxK, bf16 row-major) * Bt(NxK, bf16 row-major)^T  (+ residual)
// Block: 256 threads = 8 waves arranged WMW x WNW.
// Each wave owns a 32 x (16*NT) tile of C: 2 A-frags reused over NT B-frags
// -> 2*NT WMMAs per 32-wide K-step.
template <int WMW, int WNW, int NT, int EPI>
__global__ void __launch_bounds__(256)
gemm_bf16_kernel(const bf16_t* A, int lda, size_t strideAz,
                 const bf16_t* B, int ldb, size_t strideBz,
                 float* C, int ldc, size_t strideCz,
                 const float* R,                       // residual (EPI==1)
                 int K, int nmax) {
  A += (size_t)blockIdx.z * strideAz;
  B += (size_t)blockIdx.z * strideBz;
  C += (size_t)blockIdx.z * strideCz;
  if (EPI == 1) R += (size_t)blockIdx.z * strideCz;

  const int wave = threadIdx.x >> 5;
  const int lane = threadIdx.x & 31;
  const int wm = wave / WNW, wn = wave % WNW;
  const int m0 = (blockIdx.y * WMW + wm) * 32;
  const int n0 = (blockIdx.x * WNW + wn) * (16 * NT);
  const int row = lane & 15, hi = lane >> 4;

  const bf16_t* a0 = A + (size_t)(m0 + row) * lda + hi * 16;
  const bf16_t* a1 = A + (size_t)(m0 + 16 + row) * lda + hi * 16;
  const bf16_t* bp[NT];
#pragma unroll
  for (int j = 0; j < NT; ++j)
    bp[j] = B + (size_t)(n0 + j * 16 + row) * ldb + hi * 16;

  f32x8 acc[2][NT];
#pragma unroll
  for (int i = 0; i < 2; ++i)
#pragma unroll
    for (int j = 0; j < NT; ++j)
#pragma unroll
      for (int r = 0; r < 8; ++r) acc[i][j][r] = 0.f;

  for (int k = 0; k < K; k += 32) {
    // pull the fragment streams ahead into near cache (global_prefetch_b8)
    __builtin_prefetch(a0 + k + 256, 0, 1);
    __builtin_prefetch(bp[0] + k + 256, 0, 1);
    bf16x16 af0 = load_afrag(a0 + k);
    bf16x16 af1 = load_afrag(a1 + k);
#pragma unroll
    for (int j = 0; j < NT; ++j) {
      bf16x16 bfj = *(const bf16x16*)(bp[j] + k);
      acc[0][j] = __builtin_amdgcn_wmma_f32_16x16x32_bf16(false, af0, false, bfj,
                                                          (short)0, acc[0][j], false, false);
      acc[1][j] = __builtin_amdgcn_wmma_f32_16x16x32_bf16(false, af1, false, bfj,
                                                          (short)0, acc[1][j], false, false);
    }
  }

#pragma unroll
  for (int i = 0; i < 2; ++i) {
#pragma unroll
    for (int j = 0; j < NT; ++j) {
      const int col = n0 + j * 16 + row;
      if (col < nmax) {
        const size_t mbase = (size_t)(m0 + i * 16 + hi * 8);
#pragma unroll
        for (int r = 0; r < 8; ++r) {
          size_t off = (mbase + r) * (size_t)ldc + col;
          float v = acc[i][j][r];
          if (EPI == 1) v += R[off];
          C[off] = v;
        }
      }
    }
  }
}

// ---------------------------------------------------------------------------
// Support kernels
// ---------------------------------------------------------------------------
__global__ void __launch_bounds__(256)
embed_kernel(const int* __restrict__ ids, const float* __restrict__ temb,
             float* __restrict__ x) {
  size_t i = (size_t)blockIdx.x * 256 + threadIdx.x;
  int s = (int)(i >> 10), d = (int)(i & (D_MODEL - 1));
  x[i] = temb[(size_t)ids[s] * D_MODEL + d];
}

// fp32 KxN weight -> bf16 NxK (LDS-tiled transpose); grid=(Npad/32, Kd/32)
__global__ void __launch_bounds__(256)
wtrans_kernel(const float* __restrict__ W, bf16_t* __restrict__ Wt, int Kd, int Nd) {
  __shared__ float tile[32][33];
  const int n0 = blockIdx.x * 32, k0 = blockIdx.y * 32;
  const int tx = threadIdx.x & 31, ty = threadIdx.x >> 5;
#pragma unroll
  for (int j = 0; j < 4; ++j) {
    int k = k0 + ty + j * 8, n = n0 + tx;
    tile[ty + j * 8][tx] = (n < Nd) ? W[(size_t)k * Nd + n] : 0.f;
  }
  __syncthreads();
#pragma unroll
  for (int j = 0; j < 4; ++j) {
    int n = n0 + ty + j * 8, k = k0 + tx;
    Wt[(size_t)n * Kd + k] = (bf16_t)tile[tx][ty + j * 8];
  }
}

// rowwise layernorm of f32 x -> bf16 out; one block (256 thr) per row of 1024
__global__ void __launch_bounds__(256)
layernorm_kernel(const float* __restrict__ x, const float* __restrict__ sc,
                 const float* __restrict__ bi, bf16_t* __restrict__ out) {
  const int rowi = blockIdx.x, t = threadIdx.x;
  const float* xr = x + (size_t)rowi * D_MODEL;
  float v[4], s = 0.f, s2 = 0.f;
#pragma unroll
  for (int j = 0; j < 4; ++j) { v[j] = xr[t + j * 256]; s += v[j]; s2 += v[j] * v[j]; }
  __shared__ float sh[256];
  sh[t] = s; __syncthreads();
  for (int o = 128; o > 0; o >>= 1) { if (t < o) sh[t] += sh[t + o]; __syncthreads(); }
  float mean = sh[0] * (1.f / D_MODEL); __syncthreads();
  sh[t] = s2; __syncthreads();
  for (int o = 128; o > 0; o >>= 1) { if (t < o) sh[t] += sh[t + o]; __syncthreads(); }
  float var = sh[0] * (1.f / D_MODEL) - mean * mean;
  float inv = rsqrtf(var + 1e-5f);
  bf16_t* orow = out + (size_t)rowi * D_MODEL;
#pragma unroll
  for (int j = 0; j < 4; ++j) {
    int d = t + j * 256;
    orow[d] = (bf16_t)((v[j] - mean) * inv * sc[d] + bi[d]);
  }
}

// q/k f32 [s][h*64+dh] -> RoPE -> bf16 [h][s][dh]
__global__ void __launch_bounds__(256)
rope_kernel(const float* __restrict__ in, bf16_t* __restrict__ out) {
  size_t idx = (size_t)blockIdx.x * 256 + threadIdx.x;
  int dh = (int)(idx & (D_HEAD - 1));
  size_t hs = idx >> 6;
  int s = (int)(hs & (S_LEN - 1));
  int h = (int)(hs >> 11);
  const float* rowp = in + (size_t)s * D_MODEL + h * D_HEAD;
  float v0 = rowp[dh];
  float r = (dh < 32) ? -rowp[dh + 32] : rowp[dh - 32];
  int i = dh & 31;
  float ang = (float)s * __powf(10000.f, -(float)i * (1.f / 32.f));
  out[idx] = (bf16_t)(v0 * __cosf(ang) + r * __sinf(ang));
}

// v f32 [s][h*64+dh] -> bf16 vT [h][dh][s]  (serves directly as NxK B-operand)
__global__ void __launch_bounds__(256)
vtrans_kernel(const float* __restrict__ in, bf16_t* __restrict__ out) {
  size_t idx = (size_t)blockIdx.x * 256 + threadIdx.x;
  int s = (int)(idx & (S_LEN - 1));
  size_t r = idx >> 11;
  int dh = (int)(r & (D_HEAD - 1));
  int h = (int)(r >> 6);
  out[idx] = (bf16_t)(in[(size_t)s * D_MODEL + h * D_HEAD + dh]);
}

// causal masked softmax: scores f32 [h][q][k] -> probs bf16 ; grid=(S, H)
__global__ void __launch_bounds__(256)
attn_softmax_kernel(const float* __restrict__ scores, bf16_t* __restrict__ attn) {
  const int q = blockIdx.x, h = blockIdx.y, t = threadIdx.x;
  const float* row = scores + ((size_t)h * S_LEN + q) * S_LEN;
  bf16_t* orow = attn + ((size_t)h * S_LEN + q) * S_LEN;
  float vals[8], mx = -3.0e38f;
#pragma unroll
  for (int j = 0; j < 8; ++j) {
    int k = t + j * 256;
    float v = (k <= q) ? row[k] * 0.125f : -3.0e38f;  // 1/sqrt(64)
    vals[j] = v; mx = fmaxf(mx, v);
  }
  __shared__ float sh[256];
  sh[t] = mx; __syncthreads();
  for (int o = 128; o > 0; o >>= 1) { if (t < o) sh[t] = fmaxf(sh[t], sh[t + o]); __syncthreads(); }
  mx = sh[0]; __syncthreads();
  float sum = 0.f;
#pragma unroll
  for (int j = 0; j < 8; ++j) {
    int k = t + j * 256;
    float e = (k <= q) ? __expf(vals[j] - mx) : 0.f;
    vals[j] = e; sum += e;
  }
  sh[t] = sum; __syncthreads();
  for (int o = 128; o > 0; o >>= 1) { if (t < o) sh[t] += sh[t + o]; __syncthreads(); }
  float inv = 1.f / sh[0];
#pragma unroll
  for (int j = 0; j < 8; ++j) orow[t + j * 256] = (bf16_t)(vals[j] * inv);
}

__global__ void __launch_bounds__(256)
cvt_bf16_kernel(const float* __restrict__ in, bf16_t* __restrict__ out, size_t n) {
  size_t i = (size_t)blockIdx.x * 256 + threadIdx.x;
  if (i < n) out[i] = (bf16_t)in[i];
}

__global__ void __launch_bounds__(256)
silu_mul_kernel(const float* __restrict__ a, const float* __restrict__ b,
                bf16_t* __restrict__ g, size_t n) {
  size_t i = (size_t)blockIdx.x * 256 + threadIdx.x;
  if (i < n) {
    float x = a[i];
    float sig = 1.f / (1.f + __expf(-x));
    g[i] = (bf16_t)(x * sig * b[i]);
  }
}

// ---------------------------------------------------------------------------
// Host orchestration
// ---------------------------------------------------------------------------
extern "C" void kernel_launch(void* const* d_in, const int* in_sizes, int n_in,
                              void* d_out, int out_size, void* d_ws, size_t ws_size,
                              hipStream_t stream) {
  (void)in_sizes; (void)n_in; (void)out_size; (void)ws_size;
  const int*   ids   = (const int*)  d_in[0];
  const float* temb  = (const float*)d_in[1];
  const float* Wq    = (const float*)d_in[2];
  const float* Wk    = (const float*)d_in[3];
  const float* Wv    = (const float*)d_in[4];
  const float* Wo    = (const float*)d_in[5];
  const float* ln1s  = (const float*)d_in[6];
  const float* ln1b  = (const float*)d_in[7];
  const float* ln2s  = (const float*)d_in[8];
  const float* ln2b  = (const float*)d_in[9];
  const float* w1    = (const float*)d_in[10];
  const float* w2    = (const float*)d_in[11];
  const float* w3    = (const float*)d_in[12];
  const float* lnfs  = (const float*)d_in[13];
  const float* lnfb  = (const float*)d_in[14];
  const float* lmh   = (const float*)d_in[15];
  float* out = (float*)d_out;

  // workspace carve-up (recomputed every call -> deterministic)
  char* ws = (char*)d_ws;
  size_t off = 0;
  auto alloc = [&](size_t bytes) {
    size_t r = off; off = (off + bytes + 255) & ~(size_t)255; return r;
  };
  const size_t SD = (size_t)S_LEN * D_MODEL, SF = (size_t)S_LEN * F_FFN;
  const size_t HSS = (size_t)N_HEAD * S_LEN * S_LEN;
  float*  x    = (float*) (ws + alloc(SD * 4));
  bf16_t* hbf  = (bf16_t*)(ws + alloc(SD * 2));
  float*  qf   = (float*) (ws + alloc(SD * 4));
  float*  kf   = (float*) (ws + alloc(SD * 4));
  float*  vf   = (float*) (ws + alloc(SD * 4));
  bf16_t* qr   = (bf16_t*)(ws + alloc(SD * 2));   // [h][s][dh]
  bf16_t* kr   = (bf16_t*)(ws + alloc(SD * 2));   // [h][s][dh]
  bf16_t* vT   = (bf16_t*)(ws + alloc(SD * 2));   // [h][dh][s]
  float*  sc   = (float*) (ws + alloc(HSS * 4));  // scores
  bf16_t* at   = (bf16_t*)(ws + alloc(HSS * 2));  // probs
  float*  aof  = (float*) (ws + alloc(SD * 4));
  bf16_t* aob  = (bf16_t*)(ws + alloc(SD * 2));
  float*  f1   = (float*) (ws + alloc(SF * 4));
  float*  f3   = (float*) (ws + alloc(SF * 4));
  bf16_t* gbf  = (bf16_t*)(ws + alloc(SF * 2));
  bf16_t* wqT  = (bf16_t*)(ws + alloc((size_t)D_MODEL * D_MODEL * 2));
  bf16_t* wkT  = (bf16_t*)(ws + alloc((size_t)D_MODEL * D_MODEL * 2));
  bf16_t* wvT  = (bf16_t*)(ws + alloc((size_t)D_MODEL * D_MODEL * 2));
  bf16_t* woT  = (bf16_t*)(ws + alloc((size_t)D_MODEL * D_MODEL * 2));
  bf16_t* w1T  = (bf16_t*)(ws + alloc((size_t)F_FFN * D_MODEL * 2));
  bf16_t* w3T  = (bf16_t*)(ws + alloc((size_t)F_FFN * D_MODEL * 2));
  bf16_t* w2T  = (bf16_t*)(ws + alloc((size_t)D_MODEL * F_FFN * 2));
  bf16_t* lmT  = (bf16_t*)(ws + alloc((size_t)V_PAD * D_MODEL * 2));

  const dim3 blk(256);
  const size_t DD = (size_t)D_MODEL * D_MODEL, DF = (size_t)D_MODEL * F_FFN;

  // token embedding gather
  embed_kernel<<<dim3(SD / 256), blk, 0, stream>>>(ids, temb, x);

  // lm_head f32 (D x V) -> bf16 (Vpad x D), zero-padded rows
  wtrans_kernel<<<dim3(V_PAD / 32, D_MODEL / 32), blk, 0, stream>>>(lmh, lmT, D_MODEL, V_VOCAB);

  for (int l = 0; l < N_LAYER; ++l) {
    // --- weight convert+transpose for this layer (bf16 NxK, L2-resident) ---
    wtrans_kernel<<<dim3(32, 32), blk, 0, stream>>>(Wq + l * DD, wqT, D_MODEL, D_MODEL);
    wtrans_kernel<<<dim3(32, 32), blk, 0, stream>>>(Wk + l * DD, wkT, D_MODEL, D_MODEL);
    wtrans_kernel<<<dim3(32, 32), blk, 0, stream>>>(Wv + l * DD, wvT, D_MODEL, D_MODEL);
    wtrans_kernel<<<dim3(32, 32), blk, 0, stream>>>(Wo + l * DD, woT, D_MODEL, D_MODEL);
    wtrans_kernel<<<dim3(128, 32), blk, 0, stream>>>(w1 + l * DF, w1T, D_MODEL, F_FFN);
    wtrans_kernel<<<dim3(128, 32), blk, 0, stream>>>(w3 + l * DF, w3T, D_MODEL, F_FFN);
    wtrans_kernel<<<dim3(32, 128), blk, 0, stream>>>(w2 + l * DF, w2T, F_FFN, D_MODEL);

    // --- attention ---
    layernorm_kernel<<<dim3(S_LEN), blk, 0, stream>>>(x, ln1s + l * D_MODEL, ln1b + l * D_MODEL, hbf);
    gemm_bf16_kernel<4, 2, 4, 0><<<dim3(8, 16, 1), blk, 0, stream>>>(
        hbf, D_MODEL, 0, wqT, D_MODEL, 0, qf, D_MODEL, 0, nullptr, D_MODEL, D_MODEL);
    gemm_bf16_kernel<4, 2, 4, 0><<<dim3(8, 16, 1), blk, 0, stream>>>(
        hbf, D_MODEL, 0, wkT, D_MODEL, 0, kf, D_MODEL, 0, nullptr, D_MODEL, D_MODEL);
    gemm_bf16_kernel<4, 2, 4, 0><<<dim3(8, 16, 1), blk, 0, stream>>>(
        hbf, D_MODEL, 0, wvT, D_MODEL, 0, vf, D_MODEL, 0, nullptr, D_MODEL, D_MODEL);
    rope_kernel<<<dim3(SD / 256), blk, 0, stream>>>(qf, qr);
    rope_kernel<<<dim3(SD / 256), blk, 0, stream>>>(kf, kr);
    vtrans_kernel<<<dim3(SD / 256), blk, 0, stream>>>(vf, vT);

    // scores[h] = q[h] (S x 64) * k[h]^T ; batched over heads via grid.z
    gemm_bf16_kernel<4, 2, 4, 0><<<dim3(16, 16, N_HEAD), blk, 0, stream>>>(
        qr, D_HEAD, (size_t)S_LEN * D_HEAD,
        kr, D_HEAD, (size_t)S_LEN * D_HEAD,
        sc, S_LEN, (size_t)S_LEN * S_LEN, nullptr, D_HEAD, S_LEN);
    attn_softmax_kernel<<<dim3(S_LEN, N_HEAD), blk, 0, stream>>>(sc, at);
    // out[h] = probs[h] (S x S) * v[h] (S x 64) ; vT already NxK
    gemm_bf16_kernel<4, 2, 2, 0><<<dim3(1, 16, N_HEAD), blk, 0, stream>>>(
        at, S_LEN, (size_t)S_LEN * S_LEN,
        vT, S_LEN, (size_t)D_HEAD * S_LEN,
        aof, D_MODEL, (size_t)D_HEAD, nullptr, S_LEN, D_HEAD);
    cvt_bf16_kernel<<<dim3(SD / 256), blk, 0, stream>>>(aof, aob, SD);
    // x += out @ Wo
    gemm_bf16_kernel<4, 2, 4, 1><<<dim3(8, 16, 1), blk, 0, stream>>>(
        aob, D_MODEL, 0, woT, D_MODEL, 0, x, D_MODEL, 0, x, D_MODEL, D_MODEL);

    // --- FFN ---
    layernorm_kernel<<<dim3(S_LEN), blk, 0, stream>>>(x, ln2s + l * D_MODEL, ln2b + l * D_MODEL, hbf);
    gemm_bf16_kernel<4, 2, 4, 0><<<dim3(32, 16, 1), blk, 0, stream>>>(
        hbf, D_MODEL, 0, w1T, D_MODEL, 0, f1, F_FFN, 0, nullptr, D_MODEL, F_FFN);
    gemm_bf16_kernel<4, 2, 4, 0><<<dim3(32, 16, 1), blk, 0, stream>>>(
        hbf, D_MODEL, 0, w3T, D_MODEL, 0, f3, F_FFN, 0, nullptr, D_MODEL, F_FFN);
    silu_mul_kernel<<<dim3(SF / 256), blk, 0, stream>>>(f1, f3, gbf, SF);
    // x += g @ w2
    gemm_bf16_kernel<4, 2, 4, 1><<<dim3(8, 16, 1), blk, 0, stream>>>(
        gbf, F_FFN, 0, w2T, F_FFN, 0, x, D_MODEL, 0, x, F_FFN, D_MODEL);
  }

  // final LN + lm_head (N padded to 1408, masked store at 1400)
  layernorm_kernel<<<dim3(S_LEN), blk, 0, stream>>>(x, lnfs, lnfb, hbf);
  gemm_bf16_kernel<4, 2, 4, 0><<<dim3(V_PAD / 128, 16, 1), blk, 0, stream>>>(
      hbf, D_MODEL, 0, lmT, D_MODEL, 0, out, V_VOCAB, 0, nullptr, D_MODEL, V_VOCAB);
}
